// ConvModule_25211458027711
// MI455X (gfx1250) — matrix operand order repaired and checked by reading, not compiled, not run on hardware
//
#include <hip/hip_runtime.h>
#include <hip/hip_bf16.h>

#define BDIM   32
#define FDIM   4096
#define CDIM   512
#define HALF_F 2048
#define EPSV   1e-5f
#define LT     64     // l-columns per block tile
#define DT     256    // d-rows per block tile (8 waves x 2 sub-tiles x 16)

typedef __attribute__((ext_vector_type(16))) __bf16 v16bf;
typedef __attribute__((ext_vector_type(8)))  float  v8f;

// ---------------- math helpers (shared by stats + GEMM kernels) ----------------

__device__ __forceinline__ float sigmoidf_(float x) {
    return 1.0f / (1.0f + __expf(-x));
}

// glu[b,c,j] = (w1*h[j]+b1) * sigmoid(w1*h[j+2048]+b1),  j in [0,2048)
__device__ __forceinline__ float glu_j(const float* __restrict__ hrow,
                                       float w1c, float b1c, int j) {
    float a = fmaf(w1c, hrow[j], b1c);
    float g = fmaf(w1c, hrow[j + HALF_F], b1c);
    return a * sigmoidf_(g);
}

// depthwise(k=3,pad=1) of upsampled GLU, + dw bias. u[l] = glu[l>>1].
// Even/odd identity: u[2m] == u[2m+1], so only 2 distinct glu evals needed.
__device__ __forceinline__ float hdw_val(const float* __restrict__ hrow,
                                         float w1c, float b1c,
                                         float dw0, float dw1, float dw2, float dwb,
                                         int l) {
    int j = l >> 1;
    float uc = glu_j(hrow, w1c, b1c, j);
    float um, up;
    if (l & 1) {               // l odd: u[l-1] shares j; u[l+1] uses j+1
        um = uc;
        up = (l == FDIM - 1) ? 0.0f : glu_j(hrow, w1c, b1c, j + 1);
    } else {                   // l even: u[l+1] shares j; u[l-1] uses j-1
        up = uc;
        um = (l == 0) ? 0.0f : glu_j(hrow, w1c, b1c, j - 1);
    }
    return fmaf(dw0, um, fmaf(dw1, uc, fmaf(dw2, up, dwb)));
}

// fp32 -> bf16, round to nearest even
__device__ __forceinline__ unsigned short f2bf(float f) {
    union { float f; unsigned u; } v; v.f = f;
    unsigned r = v.u + 0x7FFFu + ((v.u >> 16) & 1u);
    return (unsigned short)(r >> 16);
}

// ---------------- kernel 1: LayerNorm over F ----------------

__global__ __launch_bounds__(256) void ln_kernel(const float* __restrict__ x,
                                                 const float* __restrict__ g,
                                                 const float* __restrict__ be,
                                                 float* __restrict__ hln) {
    __shared__ float red[256];
    const int b   = blockIdx.x;
    const int tid = threadIdx.x;
    const float* xr = x + (size_t)b * FDIM;

    float s = 0.0f, s2 = 0.0f;
    for (int j = tid; j < FDIM; j += 256) {
        float v = xr[j];
        s += v; s2 += v * v;
    }
    red[tid] = s; __syncthreads();
    for (int off = 128; off > 0; off >>= 1) {
        if (tid < off) red[tid] += red[tid + off];
        __syncthreads();
    }
    float mu = red[0] * (1.0f / FDIM);
    __syncthreads();
    red[tid] = s2; __syncthreads();
    for (int off = 128; off > 0; off >>= 1) {
        if (tid < off) red[tid] += red[tid + off];
        __syncthreads();
    }
    float var  = red[0] * (1.0f / FDIM) - mu * mu;
    float rstd = rsqrtf(var + EPSV);
    for (int j = tid; j < FDIM; j += 256) {
        hln[(size_t)b * FDIM + j] = (xr[j] - mu) * rstd * g[j] + be[j];
    }
}

// ---------------- kernel 2: per-channel BatchNorm stats, folded into scale/shift --------

__global__ __launch_bounds__(256) void stats_kernel(const float* __restrict__ hln,
                                                    const float* __restrict__ w1,
                                                    const float* __restrict__ b1,
                                                    const float* __restrict__ dw_w,
                                                    const float* __restrict__ dw_b,
                                                    const float* __restrict__ bn_g,
                                                    const float* __restrict__ bn_b,
                                                    float* __restrict__ scale,
                                                    float* __restrict__ shift) {
    __shared__ float r1[256], r2[256];
    const int c   = blockIdx.x;
    const int tid = threadIdx.x;
    const float w1c = w1[c], b1c = b1[c];
    const float dw0 = dw_w[3 * c + 0], dw1 = dw_w[3 * c + 1], dw2 = dw_w[3 * c + 2];
    const float dwb = dw_b[c];

    float s = 0.0f, s2 = 0.0f;
    for (int b = 0; b < BDIM; ++b) {
        const float* hrow = hln + (size_t)b * FDIM;
        for (int l = tid; l < FDIM; l += 256) {
            float v = hdw_val(hrow, w1c, b1c, dw0, dw1, dw2, dwb, l);
            s += v; s2 += v * v;
        }
    }
    r1[tid] = s; r2[tid] = s2; __syncthreads();
    for (int off = 128; off > 0; off >>= 1) {
        if (tid < off) { r1[tid] += r1[tid + off]; r2[tid] += r2[tid + off]; }
        __syncthreads();
    }
    if (tid == 0) {
        const float inv_n = 1.0f / (float)(BDIM * FDIM);
        float mu  = r1[0] * inv_n;
        float var = r2[0] * inv_n - mu * mu;
        float rs  = rsqrtf(var + EPSV);
        float sc  = bn_g[c] * rs;
        scale[c]  = sc;
        shift[c]  = bn_b[c] - mu * sc;
    }
}

// ---------------- kernel 3: w2 fp32 -> bf16 ----------------

__global__ __launch_bounds__(256) void cvt_kernel(const float* __restrict__ w2,
                                                  unsigned short* __restrict__ w2b) {
    int i = blockIdx.x * 256 + threadIdx.x;
    w2b[i] = f2bf(w2[i]);
}

// ---------------- kernel 4: fused activation + bf16 WMMA GEMM ----------------
// out[b,d,l] = sum_c w2[d,c] * silu(scale[c]*hdw(b,c,l)+shift[c]) + b2[d]
// Block: 256 thr (8 waves). Tile: DT=256 d-rows x LT=64 l-cols, one b.
// Each wave: 2 d sub-tiles x 4 l sub-tiles = 8 f32 16x16 accumulators.
// LDS: activation tile [LT][CDIM] bf16 (l-major so K pairs are dword-contiguous,
//      and each B fragment is two aligned 16B chunks -> 2x ds_load_b128).

__global__ __launch_bounds__(256) void gemm_kernel(const float* __restrict__ hln,
                                                   const float* __restrict__ w1,
                                                   const float* __restrict__ b1,
                                                   const float* __restrict__ dw_w,
                                                   const float* __restrict__ dw_b,
                                                   const float* __restrict__ scale,
                                                   const float* __restrict__ shift,
                                                   const unsigned short* __restrict__ w2b,
                                                   const float* __restrict__ b2,
                                                   float* __restrict__ out) {
    extern __shared__ unsigned short sB[];   // [LT][CDIM] bf16 = 64 KB
    const int l0  = blockIdx.x * LT;
    const int b   = blockIdx.y;
    const int d0  = blockIdx.z * DT;
    const int tid = threadIdx.x;
    const float* hrow = hln + (size_t)b * FDIM;

    // ---- Stage B tile: fused activation for c in [0,512), l in [l0,l0+LT) ----
    for (int c = tid; c < CDIM; c += 256) {          // 2 channels per thread
        const float w1c = w1[c], b1c = b1[c];
        const float dw0 = dw_w[3 * c + 0], dw1 = dw_w[3 * c + 1], dw2 = dw_w[3 * c + 2];
        const float dwb = dw_b[c];
        const float sc  = scale[c], sh = shift[c];
        for (int ll = 0; ll < LT; ++ll) {
            float z = hdw_val(hrow, w1c, b1c, dw0, dw1, dw2, dwb, l0 + ll);
            z = fmaf(sc, z, sh);                     // BatchNorm (folded)
            float act = z * sigmoidf_(z);            // SiLU
            sB[ll * CDIM + c] = f2bf(act);
        }
    }
    __syncthreads();

    // ---- WMMA: D(16x16 f32) += A(16x32 bf16, w2 rows) x B(32x16 bf16, act) ----
    const int wave  = tid >> 5;           // 0..7
    const int lane  = tid & 31;
    const int laneM = lane & 15;          // A row / B col / D col within tile
    const int hi    = lane >> 4;          // lane group 0/1 (ISA K split)
    const int drow0 = d0 + wave * 32 + laneM;        // d sub-tile 0 row for this lane
    const int drow1 = drow0 + 16;                    // d sub-tile 1 row

    const uint4* w2q = (const uint4*)w2b;   // 8 bf16 per uint4; row = 64 uint4
    const uint4* sBq = (const uint4*)sB;    // l-row = 64 uint4

    v8f a00 = {}, a01 = {}, a02 = {}, a03 = {};   // d sub-tile 0, l sub-tiles 0..3
    v8f a10 = {}, a11 = {}, a12 = {}, a13 = {};   // d sub-tile 1

    for (int kb = 0; kb < CDIM / 32; ++kb) {      // 16 K-steps of 32
        // A fragment: 16-bit A 16x32 layout — pair index = row*256 + kb*16 + 4*hi
        //   + v (v=0..3) and +8+v (v=4..7): two aligned 16B chunks.
        union { uint4 q[2]; v16bf v; } A0, A1;
        const int a0 = drow0 * 64 + kb * 4 + hi;  // uint4 index of chunk0
        const int a1 = drow1 * 64 + kb * 4 + hi;
        A0.q[0] = w2q[a0];  A0.q[1] = w2q[a0 + 2];
        A1.q[0] = w2q[a1];  A1.q[1] = w2q[a1 + 2];

        // B fragments: 16-bit B 32x16 layout — pair index = l*256 + kb*16 + 4*hi
        //   + v (v=0..3) and +8+v (v=4..7): two aligned 16B LDS chunks (ds_load_b128 x2).
#pragma unroll
        for (int lt = 0; lt < 4; ++lt) {
            union { uint4 q[2]; v16bf v; } Bm;
            const int bq = (lt * 16 + laneM) * 64 + kb * 4 + hi;
            Bm.q[0] = sBq[bq];  Bm.q[1] = sBq[bq + 2];

            if (lt == 0) {
                a00 = __builtin_amdgcn_wmma_f32_16x16x32_bf16(false, A0.v, false, Bm.v, (short)0, a00, false, false);
                a10 = __builtin_amdgcn_wmma_f32_16x16x32_bf16(false, A1.v, false, Bm.v, (short)0, a10, false, false);
            }
            if (lt == 1) {
                a01 = __builtin_amdgcn_wmma_f32_16x16x32_bf16(false, A0.v, false, Bm.v, (short)0, a01, false, false);
                a11 = __builtin_amdgcn_wmma_f32_16x16x32_bf16(false, A1.v, false, Bm.v, (short)0, a11, false, false);
            }
            if (lt == 2) {
                a02 = __builtin_amdgcn_wmma_f32_16x16x32_bf16(false, A0.v, false, Bm.v, (short)0, a02, false, false);
                a12 = __builtin_amdgcn_wmma_f32_16x16x32_bf16(false, A1.v, false, Bm.v, (short)0, a12, false, false);
            }
            if (lt == 3) {
                a03 = __builtin_amdgcn_wmma_f32_16x16x32_bf16(false, A0.v, false, Bm.v, (short)0, a03, false, false);
                a13 = __builtin_amdgcn_wmma_f32_16x16x32_bf16(false, A1.v, false, Bm.v, (short)0, a13, false, false);
            }
        }
    }

    // ---- Epilogue: D layout — VGPR r holds row M = r + 8*hi, col N = laneM ----
#pragma unroll
    for (int r = 0; r < 8; ++r) {
        const int dA = d0 + wave * 32 + r + 8 * hi;        // d sub-tile 0
        const int dB = dA + 16;                            // d sub-tile 1
        const float biasA = b2[dA];
        const float biasB = b2[dB];
        float* orowA = out + ((size_t)b * CDIM + dA) * FDIM + l0;
        float* orowB = out + ((size_t)b * CDIM + dB) * FDIM + l0;
        orowA[0 * 16 + laneM] = a00[r] + biasA;
        orowA[1 * 16 + laneM] = a01[r] + biasA;
        orowA[2 * 16 + laneM] = a02[r] + biasA;
        orowA[3 * 16 + laneM] = a03[r] + biasA;
        orowB[0 * 16 + laneM] = a10[r] + biasB;
        orowB[1 * 16 + laneM] = a11[r] + biasB;
        orowB[2 * 16 + laneM] = a12[r] + biasB;
        orowB[3 * 16 + laneM] = a13[r] + biasB;
    }
}

// ---------------- host-side launch ----------------

extern "C" void kernel_launch(void* const* d_in, const int* in_sizes, int n_in,
                              void* d_out, int out_size, void* d_ws, size_t ws_size,
                              hipStream_t stream) {
    const float* x    = (const float*)d_in[0];
    const float* ln_g = (const float*)d_in[1];
    const float* ln_b = (const float*)d_in[2];
    const float* w1   = (const float*)d_in[3];
    const float* b1   = (const float*)d_in[4];
    const float* dw_w = (const float*)d_in[5];
    const float* dw_b = (const float*)d_in[6];
    const float* bn_g = (const float*)d_in[7];
    const float* bn_b = (const float*)d_in[8];
    const float* w2   = (const float*)d_in[9];
    const float* b2   = (const float*)d_in[10];
    float* out = (float*)d_out;

    // workspace: hln [B,F] f32 | scale [C] | shift [C] | w2 bf16 [C,C]
    float* hln            = (float*)d_ws;
    float* scale          = hln + (size_t)BDIM * FDIM;
    float* shift          = scale + CDIM;
    unsigned short* w2b   = (unsigned short*)(shift + CDIM);

    ln_kernel<<<BDIM, 256, 0, stream>>>(x, ln_g, ln_b, hln);
    cvt_kernel<<<(CDIM * CDIM) / 256, 256, 0, stream>>>(w2, w2b);
    stats_kernel<<<CDIM, 256, 0, stream>>>(hln, w1, b1, dw_w, dw_b, bn_g, bn_b, scale, shift);

    dim3 grid(FDIM / LT, BDIM, CDIM / DT);
    size_t smem = (size_t)LT * CDIM * sizeof(unsigned short);   // 64 KB
    gemm_kernel<<<grid, 256, smem, stream>>>(hln, w1, b1, dw_w, dw_b,
                                             scale, shift, w2b, b2, out);
}
